// SpatialBranch_drg_84610855731250
// MI455X (gfx1250) — compile-verified, matrix operand order checked by reading, not run on hardware
//
#include <hip/hip_runtime.h>

// ---------------------------------------------------------------------------
// SpatialBranch (DRG) for MI455X / gfx1250, wave32 + WMMA + Tensor Data Mover.
// pair-enum -> weight repack (f16) -> fused(SAT conv1, conv2 WMMA, pool)
//           -> linear WMMA GEMM with TDM double-buffered LDS staging.
// ---------------------------------------------------------------------------

typedef __attribute__((ext_vector_type(16))) _Float16 v16h;
typedef __attribute__((ext_vector_type(8)))  float    v8f;
typedef __attribute__((ext_vector_type(4)))  unsigned v4u;
typedef __attribute__((ext_vector_type(8)))  int      v8i_t;
typedef __attribute__((ext_vector_type(4)))  int      v4i_t;

union Frag16 {            // 8 VGPRs = 16 f16 (A or B fragment of 16x16x32 WMMA)
  v16h v;
  _Float16 h[16];
  unsigned u[8];
};

#define B_BATCH 8
#define NMAX 24
#define MAXM 2208           // 8 * C(24,2)
#define LIN_K 5408
#define LIN_N 4800

// ws layout (bytes):
#define WS_BI   0u
#define WS_II   8832u
#define WS_JI   17664u
#define WS_W2H  26624u        // f16[51200] conv2 weights, order [oc][(ky*5+kx)][ic]
#define WS_X    131072u       // f16[MAXM][5408] activations
#define WS_WT16 24012800u     // f16[4800][5408] transposed linear weights
#define WS_FULL_BYTES 75929600u

// ---------------------------------------------------------------------------
// TDM: issue a 2D f16 tile load (tile0 contiguous elems x tile1 rows) to LDS.
// D# packing per CDNA5 ISA §8.3/8.4. Uniform args only; issue from one wave.
// ---------------------------------------------------------------------------
__device__ __forceinline__ void tdm_load_2d_f16(unsigned lds_byte_addr,
                                                unsigned long long ga_bytes,
                                                unsigned tile0, unsigned tile1,
                                                unsigned long long stride0_elems,
                                                unsigned td0, unsigned td1) {
  v4u g0;
  g0.x = 1u;                                              // count=1, user D#
  g0.y = lds_byte_addr;                                   // lds_addr [63:32]
  g0.z = (unsigned)(ga_bytes & 0xffffffffull);            // global_addr lo
  g0.w = (unsigned)((ga_bytes >> 32) & 0x01ffffffull)     // global_addr hi
         | (2u << 30);                                    // type=2 (image)
  union { unsigned w[8]; v8i_t v; } g1;
  g1.w[0] = (1u << 16);                                   // data_size=1 (2B)
  g1.w[1] = (td0 & 0xffffu) << 16;                        // tensor_dim0[15:0]
  g1.w[2] = (td0 >> 16) | ((td1 & 0xffffu) << 16);        // td0 hi | td1 lo
  g1.w[3] = (td1 >> 16) | ((tile0 & 0xffffu) << 16);      // td1 hi | tile_dim0
  g1.w[4] = (tile1 & 0xffffu);                            // tile_dim1 (tile2=0)
  g1.w[5] = (unsigned)(stride0_elems & 0xffffffffull);    // dim0_stride lo
  g1.w[6] = (unsigned)((stride0_elems >> 32) & 0xffffull);// dim0_stride hi
  g1.w[7] = 0;
  v4i_t z4 = {0, 0, 0, 0};
#if __clang_major__ >= 23
  v8i_t z8 = {0, 0, 0, 0, 0, 0, 0, 0};
  __builtin_amdgcn_tensor_load_to_lds(g0, g1.v, z4, z4, z8, 0);
#else
  __builtin_amdgcn_tensor_load_to_lds(g0, g1.v, z4, z4, 0);
#endif
}

// ---------------------------------------------------------------------------
// Kernel 0: enumerate ragged upper-triangular pairs; write slicing output.
// ---------------------------------------------------------------------------
__global__ void build_pairs_kernel(const int* __restrict__ num_obj,
                                   int* __restrict__ bi, int* __restrict__ ii,
                                   int* __restrict__ ji, float* __restrict__ slic,
                                   int M) {
  __shared__ int off[B_BATCH + 1];
  if (threadIdx.x == 0) {
    int acc = 0;
    for (int b = 0; b < B_BATCH; ++b) {
      off[b] = acc;
      int n = num_obj[b];
      acc += n * (n - 1) / 2;
    }
    off[B_BATCH] = acc;
  }
  __syncthreads();
  for (int p = threadIdx.x; p < M; p += blockDim.x) {
    int b = 0;
    while (b < B_BATCH - 1 && p >= off[b + 1]) ++b;
    int l = p - off[b];
    int n = num_obj[b];
    int i = 0, rem = l;
    while (rem >= n - 1 - i) { rem -= n - 1 - i; ++i; }   // triu_indices order
    int j = i + 1 + rem;
    bi[p] = b; ii[p] = i; ji[p] = j;
    slic[p * 3 + 0] = (float)b;
    slic[p * 3 + 1] = (float)i;
    slic[p * 3 + 2] = (float)j;
  }
}

// ---------------------------------------------------------------------------
// Kernel 1a: conv2 weights f32 [32][64][5][5] -> f16 ws, order [oc][g][ic],
//            g = ky*5+kx  (matches channel-contiguous im2col K order).
// ---------------------------------------------------------------------------
__global__ __launch_bounds__(256)
void convert_w2_kernel(const float* __restrict__ c2w, _Float16* __restrict__ w2h) {
  int idx = blockIdx.x * 256 + threadIdx.x;     // 51200 total
  if (idx < 51200) {
    int oc = idx / 1600;
    int r  = idx - oc * 1600;
    int g  = r >> 6;                            // 0..24
    int ic = r & 63;
    w2h[idx] = (_Float16)c2w[(oc * 64 + ic) * 25 + g];
  }
}

// ---------------------------------------------------------------------------
// Kernel 1b: lin_w f32 [5408][4800] -> f16 transposed [4800][5408] in ws.
// ---------------------------------------------------------------------------
__global__ __launch_bounds__(256)
void transpose_lin_w_kernel(const float* __restrict__ W, _Float16* __restrict__ Wt) {
  __shared__ _Float16 tile[32 * 33];
  int k0 = blockIdx.x * 32;                     // 169 blocks
  int n0 = blockIdx.y * 32;                     // 150 blocks
  #pragma unroll
  for (int e = 0; e < 4; ++e) {
    int idx = e * 256 + threadIdx.x;
    int kr = idx >> 5, nc = idx & 31;
    tile[nc * 33 + kr] = (_Float16)W[(size_t)(k0 + kr) * LIN_N + n0 + nc];
  }
  __syncthreads();
  #pragma unroll
  for (int e = 0; e < 4; ++e) {
    int idx = e * 256 + threadIdx.x;
    int nr = idx >> 5, kc = idx & 31;
    Wt[(size_t)(n0 + nr) * LIN_K + k0 + kc] = tile[nr * 33 + kc];
  }
}

// ---------------------------------------------------------------------------
// Kernel 2: fused per-pair stage. One workgroup (256 threads = 8 waves) / pair.
// Dynamic LDS carve (279,680 B), base offset 0 (no static LDS):
//   h1t : conv1+pool acts  [30*30][64] f16  @0        115200 B  (chan-contig)
//   w2t : conv2 weights    [32][1600] f16   @115200   102400 B  (TDM-loaded)
//   y2  : conv2 relu out   [32][676]  f16   @217600    43264 B
//   sat : conv1 kernel SAT [64][2][6][6] f32 @260864   18432 B
//   cb1 : conv1 bias       [64] f32         @279296      256 B
//   cb2 : conv2 bias       [32] f32         @279552      128 B
// ---------------------------------------------------------------------------
#define FUSED_SMEM_BYTES 279680
#define LDS_OFF_W2T 115200u

__global__ __launch_bounds__(256)
void fused_pair_kernel(const float* __restrict__ bboxes,
                       const int* __restrict__ bi, const int* __restrict__ ii,
                       const int* __restrict__ ji,
                       const float* __restrict__ c1w, const float* __restrict__ c1b,
                       const _Float16* __restrict__ w2h, const float* __restrict__ c2b,
                       _Float16* __restrict__ X, int M) {
  extern __shared__ char smem[];
  _Float16* h1t = (_Float16*)smem;          // 57600 halves, [pos][ic]
  _Float16* w2t = h1t + 57600;              // 51200 halves, [oc][g*64+ic]
  _Float16* y2  = w2t + 51200;              // 21632 halves
  float* sat = (float*)(y2 + 21632);        // 128 * 36 f32
  float* cb1 = sat + 128 * 36;              // 64 f32
  float* cb2 = cb1 + 64;                    // 32 f32

  const int p   = blockIdx.x;
  const int tid = threadIdx.x;
  const int lane = tid & 31;
  const int wave = tid >> 5;

  // ---- TDM: pull 100KB of conv2 weights into LDS, overlapped with conv1 ---
  if (wave == 0) {
    tdm_load_2d_f16(LDS_OFF_W2T, (unsigned long long)(size_t)w2h,
                    /*tile0*/ 51200u, /*tile1*/ 0u, /*stride*/ 0ull,
                    /*td0*/ 51200u, /*td1*/ 1u);
  }

  // ---- Phase 0a: box geometry -> integer mask bounds (all in [0,63]) ------
  int iX1[2], iX2[2], iY1[2], iY2[2];
  {
    int b = bi[p], i = ii[p], j = ji[p];
    const float* A  = bboxes + ((size_t)(b * NMAX) + i) * 4;
    const float* Bb = bboxes + ((size_t)(b * NMAX) + j) * 4;
    float ax1 = A[0],  ay1 = A[1],  ax2 = A[2],  ay2 = A[3];
    float bx1 = Bb[0], by1 = Bb[1], bx2 = Bb[2], by2 = Bb[3];
    float u1 = fminf(ax1, bx1), v1 = fminf(ay1, by1);
    float u2 = fmaxf(ax2, bx2), v2 = fmaxf(ay2, by2);
    float w_ = u2 - u1, h_ = v2 - v1;
    float scale = 63.0f / fmaxf(fmaxf(w_, h_), 1e-6f);
    float offx = (h_ >= w_) ? (63.0f - w_ * scale) * 0.5f : 0.0f;
    float offy = (w_ > h_)  ? (63.0f - h_ * scale) * 0.5f : 0.0f;
    iX1[0] = (int)rintf((ax1 - u1) * scale + offx);
    iX1[1] = (int)rintf((bx1 - u1) * scale + offx);
    iX2[0] = (int)rintf((ax2 - u1) * scale + offx);
    iX2[1] = (int)rintf((bx2 - u1) * scale + offx);
    iY1[0] = (int)rintf((ay1 - v1) * scale + offy);
    iY1[1] = (int)rintf((by1 - v1) * scale + offy);
    iY2[0] = (int)rintf((ay2 - v1) * scale + offy);
    iY2[1] = (int)rintf((by2 - v1) * scale + offy);
  }

  // ---- Phase 0b: conv1 kernel summed-area tables + biases into LDS --------
  // sat[(oc*2+c)*36 + j*6 + l] = sum_{ky<j, kx<l} w1[oc][c][ky][kx]
  if (tid < 128) {
    const float* wsrc = c1w + tid * 25;       // tid = oc*2+c
    float* S = sat + tid * 36;
    #pragma unroll
    for (int l = 0; l < 6; ++l) S[l] = 0.0f;
    #pragma unroll
    for (int jr = 1; jr < 6; ++jr) {
      float rowacc = 0.0f;
      S[jr * 6] = 0.0f;
      #pragma unroll
      for (int l = 1; l < 6; ++l) {
        rowacc += wsrc[(jr - 1) * 5 + (l - 1)];
        S[jr * 6 + l] = S[(jr - 1) * 6 + l] + rowacc;
      }
    }
  }
  if (tid < 64) cb1[tid] = c1b[tid];
  if (tid < 32) cb2[tid] = c2b[tid];
  __syncthreads();

  // ---- Phase 1: conv1+ReLU+pool via SAT -> h1t[pos][ic] (chan-contig) -----
  for (int idx = tid; idx < 64 * 900; idx += 256) {
    int oc = idx & 63;
    int rr = idx >> 6;                        // pooled pos 0..899
    int py = rr / 30;
    int px = rr - py * 30;
    const float* S0 = sat + (oc * 2) * 36;
    const float* S1 = S0 + 36;
    float bias = cb1[oc];
    float best = -1e30f;
    #pragma unroll
    for (int dy = 0; dy < 2; ++dy) {
      #pragma unroll
      for (int dx = 0; dx < 2; ++dx) {
        int y = 2 * py + dy, x = 2 * px + dx;
        float s = bias;
        #pragma unroll
        for (int c = 0; c < 2; ++c) {
          int xlo = iX1[c] - x; xlo = xlo < 0 ? 0 : xlo;
          int xhi = iX2[c] - x; xhi = xhi > 4 ? 4 : xhi;
          int ylo = iY1[c] - y; ylo = ylo < 0 ? 0 : ylo;
          int yhi = iY2[c] - y; yhi = yhi > 4 ? 4 : yhi;
          if (xhi >= xlo && yhi >= ylo) {
            const float* S = c ? S1 : S0;
            s += S[(yhi + 1) * 6 + xhi + 1] - S[ylo * 6 + xhi + 1]
               - S[(yhi + 1) * 6 + xlo]     + S[ylo * 6 + xlo];
          }
        }
        best = fmaxf(best, s);
      }
    }
    h1t[rr * 64 + oc] = (_Float16)fmaxf(best, 0.0f);   // pool(relu)==relu(pool)
  }
  if (wave == 0) __builtin_amdgcn_s_wait_tensorcnt((short)0);  // w2t landed
  __syncthreads();

  // ---- Phase 2: conv2 (64->32, 5x5) as WMMA GEMM over im2col -------------
  // rows = 676 positions, cols = 32 oc, K = 50 steps of 32 (k = g*64 + ic).
  {
    const int ocTile = wave & 1;
    const int khalf  = (lane >> 4) << 3;       // lanes 16-31: K offset +8
    const int colOc  = ocTile * 16 + (lane & 15);
    const float bias2 = cb2[colOc];
    const _Float16* brow0 = w2t + colOc * 1600;
    for (int t = (wave >> 1); t < 43; t += 4) {
      int pos = t * 16 + (lane & 15);
      int posc = pos < 676 ? pos : 675;        // clamp pad rows (not stored)
      int oy = posc / 26;
      int ox = posc - oy * 26;
      v8f acc = {};
      for (int ks = 0; ks < 50; ++ks) {
        int g   = ks >> 1;
        int ky  = g / 5;
        int kx  = g - ky * 5;
        int icb = (ks & 1) << 5;
        const _Float16* arow = h1t + ((oy + ky) * 30 + (ox + kx)) * 64 + icb;
        const _Float16* brow = brow0 + g * 64 + icb;
        Frag16 a, bf;
        #pragma unroll
        for (int jj = 0; jj < 8; ++jj) {
          int k0 = (jj < 4 ? 2 * jj : 2 * jj + 8) + khalf;   // ISA K striping
          a.u[jj]  = *(const unsigned*)(arow + k0);          // ds_load_b32
          bf.u[jj] = *(const unsigned*)(brow + k0);
        }
        acc = __builtin_amdgcn_wmma_f32_16x16x32_f16(
            false, a.v, false, bf.v, (short)0, acc, false, false);
      }
      int mbase = t * 16 + ((lane >> 4) << 3);
      _Float16* yrow = y2 + colOc * 676;
      if (t < 42) {                            // wave-uniform: no exec churn
        #pragma unroll
        for (int r = 0; r < 8; ++r)
          yrow[mbase + r] = (_Float16)fmaxf(acc[r] + bias2, 0.0f);
      } else {
        #pragma unroll
        for (int r = 0; r < 8; ++r) {
          int m = mbase + r;
          if (m < 676) yrow[m] = (_Float16)fmaxf(acc[r] + bias2, 0.0f);
        }
      }
    }
  }
  __syncthreads();

  // ---- Phase 3: 2x2 maxpool 26x26 -> 13x13, flatten NCHW -> X[p][5408] ----
  for (int idx = tid; idx < LIN_K; idx += 256) {
    int oc = idx / 169;
    int rr = idx - oc * 169;
    int py = rr / 13;
    int px = rr - py * 13;
    int base = oc * 676 + (2 * py) * 26 + 2 * px;
    float m = fmaxf(fmaxf((float)y2[base], (float)y2[base + 1]),
                    fmaxf((float)y2[base + 26], (float)y2[base + 27]));
    X[(size_t)p * LIN_K + idx] = (_Float16)m;
  }
}

// ---------------------------------------------------------------------------
// Kernel 3a (primary): linear GEMM, TDM double-buffered staging.
// out[M,4800] = relu(X[M,5408] @ W + b); W pre-transposed f16 [4800][5408].
// Block = 128 threads (4 waves); tile 64(M) x 64(N); K: 169 steps of 32.
// Dynamic LDS 16KB @0: Xt[2][64*32] then Wt[2][64*32].
// ---------------------------------------------------------------------------
__global__ __launch_bounds__(128)
void linear_wmma_tdm_kernel(const _Float16* __restrict__ X,
                            const _Float16* __restrict__ Wt,
                            const float* __restrict__ lb,
                            float* __restrict__ out, int M) {
  extern __shared__ char lsm[];
  _Float16* XtBuf = (_Float16*)lsm;            // 2 x 2048 halves
  _Float16* WtBuf = (_Float16*)(lsm + 8192);   // 2 x 2048 halves

  const int tid  = threadIdx.x;
  const int lane = tid & 31;
  const int wave = tid >> 5;
  const int m0 = blockIdx.x * 64;
  const int n0 = blockIdx.y * 64;
  const int khalf = (lane >> 4) << 3;
  const unsigned rowsLeft = (unsigned)(M - m0);   // OOB rows -> TDM zero-fill

  v8f acc[4] = {};

  if (wave == 0) {                               // prologue: k-step 0 -> buf 0
    tdm_load_2d_f16(0u, (unsigned long long)(size_t)(X + (size_t)m0 * LIN_K),
                    32u, 64u, (unsigned long long)LIN_K, 32u, rowsLeft);
    tdm_load_2d_f16(8192u, (unsigned long long)(size_t)(Wt + (size_t)n0 * LIN_K),
                    32u, 64u, (unsigned long long)LIN_K, 32u, 64u);
  }

  for (int kk = 0; kk < LIN_K / 32; ++kk) {
    const int buf = kk & 1;
    if (wave == 0) {
      if (kk + 1 < LIN_K / 32) {
        const int nb = buf ^ 1;
        tdm_load_2d_f16((unsigned)(nb * 4096),
                        (unsigned long long)(size_t)(X + (size_t)m0 * LIN_K + (kk + 1) * 32),
                        32u, 64u, (unsigned long long)LIN_K, 32u, rowsLeft);
        tdm_load_2d_f16((unsigned)(8192 + nb * 4096),
                        (unsigned long long)(size_t)(Wt + (size_t)n0 * LIN_K + (kk + 1) * 32),
                        32u, 64u, (unsigned long long)LIN_K, 32u, 64u);
        __builtin_amdgcn_s_wait_tensorcnt((short)2);  // current pair landed
      } else {
        __builtin_amdgcn_s_wait_tensorcnt((short)0);
      }
    }
    __syncthreads();

    const _Float16* Xb = XtBuf + buf * 2048;
    const _Float16* Wb = WtBuf + buf * 2048;
    Frag16 a;
    int mrow = (wave << 4) + (lane & 15);
    #pragma unroll
    for (int jj = 0; jj < 8; ++jj) {
      int k0 = (jj < 4 ? 2 * jj : 2 * jj + 8) + khalf;
      a.u[jj] = *(const unsigned*)(Xb + mrow * 32 + k0);
    }
    #pragma unroll
    for (int s = 0; s < 4; ++s) {
      Frag16 bf;
      int coll = (s << 4) + (lane & 15);
      #pragma unroll
      for (int jj = 0; jj < 8; ++jj) {
        int k0 = (jj < 4 ? 2 * jj : 2 * jj + 8) + khalf;
        bf.u[jj] = *(const unsigned*)(Wb + coll * 32 + k0);
      }
      acc[s] = __builtin_amdgcn_wmma_f32_16x16x32_f16(
          false, a.v, false, bf.v, (short)0, acc[s], false, false);
    }
    __syncthreads();
  }

  #pragma unroll
  for (int s = 0; s < 4; ++s) {
    int col = n0 + (s << 4) + (lane & 15);
    float bias = lb[col];
    #pragma unroll
    for (int r = 0; r < 8; ++r) {
      int row = m0 + (wave << 4) + r + ((lane >> 4) << 3);
      if (row < M)
        out[(size_t)row * LIN_N + col] = fmaxf(acc[s][r] + bias, 0.0f);
    }
  }
}

// ---------------------------------------------------------------------------
// Kernel 3b (fallback, small ws): in-flight f32->f16 conversion GEMM.
// ---------------------------------------------------------------------------
__global__ __launch_bounds__(128)
void linear_wmma_kernel(const _Float16* __restrict__ X,
                        const float* __restrict__ W,
                        const float* __restrict__ lb,
                        float* __restrict__ out, int M) {
  __shared__ _Float16 Xt[64 * 32];
  __shared__ _Float16 Wts[64 * 32];

  const int tid  = threadIdx.x;
  const int lane = tid & 31;
  const int wave = tid >> 5;
  const int m0 = blockIdx.x * 64;
  const int n0 = blockIdx.y * 64;
  const int khalf = (lane >> 4) << 3;

  v8f acc[4] = {};

  for (int kk = 0; kk < LIN_K / 32; ++kk) {
    __syncthreads();
    {
      int row = tid >> 1;
      int kb  = (tid & 1) * 16;
      int srcRow = m0 + row;
      if (srcRow >= M) srcRow = M - 1;
      const uint4* s = (const uint4*)(X + (size_t)srcRow * LIN_K + kk * 32 + kb);
      uint4* d = (uint4*)(Xt + row * 32 + kb);
      d[0] = s[0];
      d[1] = s[1];
    }
    #pragma unroll
    for (int e = 0; e < 16; ++e) {
      int idx = e * 128 + tid;
      int col = idx & 63;
      int kl  = idx >> 6;
      Wts[col * 32 + kl] = (_Float16)W[(size_t)(kk * 32 + kl) * LIN_N + n0 + col];
    }
    if (kk + 1 < LIN_K / 32)
      __builtin_prefetch(W + (size_t)((kk + 1) * 32) * LIN_N + n0 + lane * 16, 0, 0);
    __syncthreads();

    Frag16 a;
    int mrow = (wave << 4) + (lane & 15);
    #pragma unroll
    for (int jj = 0; jj < 8; ++jj) {
      int k0 = (jj < 4 ? 2 * jj : 2 * jj + 8) + khalf;
      a.u[jj] = *(const unsigned*)(Xt + mrow * 32 + k0);
    }
    #pragma unroll
    for (int s = 0; s < 4; ++s) {
      Frag16 bf;
      int coll = (s << 4) + (lane & 15);
      #pragma unroll
      for (int jj = 0; jj < 8; ++jj) {
        int k0 = (jj < 4 ? 2 * jj : 2 * jj + 8) + khalf;
        bf.u[jj] = *(const unsigned*)(Wts + coll * 32 + k0);
      }
      acc[s] = __builtin_amdgcn_wmma_f32_16x16x32_f16(
          false, a.v, false, bf.v, (short)0, acc[s], false, false);
    }
  }

  #pragma unroll
  for (int s = 0; s < 4; ++s) {
    int col = n0 + (s << 4) + (lane & 15);
    float bias = lb[col];
    #pragma unroll
    for (int r = 0; r < 8; ++r) {
      int row = m0 + (wave << 4) + r + ((lane >> 4) << 3);
      if (row < M)
        out[(size_t)row * LIN_N + col] = fmaxf(acc[s][r] + bias, 0.0f);
    }
  }
}

// ---------------------------------------------------------------------------
// Host launcher
// ---------------------------------------------------------------------------
extern "C" void kernel_launch(void* const* d_in, const int* in_sizes, int n_in,
                              void* d_out, int out_size, void* d_ws, size_t ws_size,
                              hipStream_t stream) {
  const float* bboxes  = (const float*)d_in[0];
  const int*   num_obj = (const int*)d_in[1];
  const float* c1w     = (const float*)d_in[2];
  const float* c1b     = (const float*)d_in[3];
  const float* c2w     = (const float*)d_in[4];
  const float* c2b     = (const float*)d_in[5];
  const float* lw      = (const float*)d_in[6];
  const float* lb      = (const float*)d_in[7];
  float* out = (float*)d_out;

  const int M = out_size / 4803;               // out = M*4800 feat + M*3 slicing
  if (M <= 0) return;

  char* ws = (char*)d_ws;
  int* bi = (int*)(ws + WS_BI);
  int* ii = (int*)(ws + WS_II);
  int* ji = (int*)(ws + WS_JI);
  _Float16* w2h  = (_Float16*)(ws + WS_W2H);
  _Float16* X    = (_Float16*)(ws + WS_X);
  _Float16* Wt16 = (_Float16*)(ws + WS_WT16);
  float* slic = out + (size_t)M * LIN_N;
  const bool full = ws_size >= (size_t)WS_FULL_BYTES;   // constant per harness

  build_pairs_kernel<<<1, 256, 0, stream>>>(num_obj, bi, ii, ji, slic, M);
  convert_w2_kernel<<<200, 256, 0, stream>>>(c2w, w2h);
  if (full)
    transpose_lin_w_kernel<<<dim3(LIN_K / 32, LIN_N / 32), 256, 0, stream>>>(lw, Wt16);

  fused_pair_kernel<<<M, 256, FUSED_SMEM_BYTES, stream>>>(
      bboxes, bi, ii, ji, c1w, c1b, w2h, c2b, X, M);

  dim3 ggrid((M + 63) / 64, LIN_N / 64);
  if (full)
    linear_wmma_tdm_kernel<<<ggrid, 128, 16384, stream>>>(X, Wt16, lb, out, M);
  else
    linear_wmma_kernel<<<ggrid, 128, 0, stream>>>(X, lw, lb, out, M);
}
